// QLSTMCell_10007273800256
// MI455X (gfx1250) — matrix operand order repaired and checked by reading, not compiled, not run on hardware
//
#include <hip/hip_runtime.h>

// Problem constants: B=4096, IN=H=2048, gates=4 (f,i,o,g), 4H=8192.
#define B_DIM  4096
#define K_DIM  2048
#define H_DIM  2048

typedef __attribute__((ext_vector_type(16))) __bf16       v16bf;
typedef __attribute__((ext_vector_type(8)))  float        v8f;
typedef __attribute__((ext_vector_type(4)))  unsigned int u32x4;
typedef __attribute__((ext_vector_type(8)))  int          i32x8;
typedef __attribute__((ext_vector_type(4)))  int          i32x4;

union FragBF {
  v16bf v;
  u32x4 u[2];
};

// ---- helpers ----------------------------------------------------------

// Pack two f32 into one dword of two bf16 (round-to-nearest-even).
__device__ __forceinline__ unsigned int pack_bf16_2(float a, float b) {
  union { float f; unsigned int u; } ua, ub;
  ua.f = a; ub.f = b;
  unsigned int ra = ua.u + 0x7FFFu + ((ua.u >> 16) & 1u);
  unsigned int rb = ub.u + 0x7FFFu + ((ub.u >> 16) & 1u);
  return (ra >> 16) | (rb & 0xFFFF0000u);
}

__device__ __forceinline__ float hsig(float x) {
  return fminf(fmaxf(x * 0.16666667f + 0.5f, 0.0f), 1.0f);
}
__device__ __forceinline__ float htanh(float x) {
  return fminf(fmaxf(x, -1.0f), 1.0f);
}

// LDS generic address: low 32 bits are the in-allocation byte offset (ISA 10.2).
__device__ __forceinline__ unsigned lds_off(const void* p) {
  return (unsigned)(unsigned long long)p;
}

// Issue one TDM tensor_load_to_lds. Tensor described per CDNA5 ISA:
//   up-to-3D tile: dim0 = contiguous k, dim1 = rows, dim2 = gate slices.
// All dims/strides in data_size (=2B) units; addresses in bytes.
__device__ __forceinline__ void tdm_load(unsigned long long gaddr_bytes,
                                         unsigned lds_addr,
                                         unsigned tensor_d0, unsigned tensor_d1,
                                         unsigned tensor_d2,
                                         unsigned tile_d0, unsigned tile_d1,
                                         unsigned tile_d2,
                                         unsigned long long stride0,
                                         unsigned long long stride1) {
  u32x4 g0 = {0u, 0u, 0u, 0u};
  g0[0] = 1u;                                    // count=1, user mode, no gather
  g0[1] = lds_addr;                              // lds_addr [63:32]
  g0[2] = (unsigned)gaddr_bytes;                 // global_addr [95:64]
  g0[3] = ((unsigned)(gaddr_bytes >> 32) & 0x1FFFFFFu) | (2u << 30);  // addr[56:32] | type=2

  i32x8 g1 = {0, 0, 0, 0, 0, 0, 0, 0};
  g1[0] = (int)(1u << 16);                       // data_size = 2 bytes (code 1)
  g1[1] = (int)((tensor_d0 & 0xFFFFu) << 16);                                  // dim0[15:0]
  g1[2] = (int)(((tensor_d0 >> 16) & 0xFFFFu) | ((tensor_d1 & 0xFFFFu) << 16)); // dim0[31:16]|dim1[15:0]
  g1[3] = (int)(((tensor_d1 >> 16) & 0xFFFFu) | (tile_d0 << 16));               // dim1[31:16]|tile0
  g1[4] = (int)(tile_d1 | (tile_d2 << 16));                                     // tile1 | tile2
  g1[5] = (int)(unsigned)(stride0 & 0xFFFFFFFFull);                             // stride0[31:0]
  g1[6] = (int)(((unsigned)(stride0 >> 32) & 0xFFFFu) |
                (((unsigned)stride1 & 0xFFFFu) << 16));                         // s0[47:32]|s1[15:0]
  g1[7] = (int)(unsigned)(stride1 >> 16);                                       // stride1[47:16]

  i32x4 g2 = {0, 0, 0, 0};
  g2[0] = (int)tensor_d2;                        // tensor_dim2 (0 for 2D)
  i32x4 g3 = {0, 0, 0, 0};

#if defined(__clang_major__) && (__clang_major__ >= 23)
  i32x8 g4 = {0, 0, 0, 0, 0, 0, 0, 0};
  __builtin_amdgcn_tensor_load_to_lds(g0, g1, g2, g3, g4, 0);
#else
  __builtin_amdgcn_tensor_load_to_lds(g0, g1, g2, g3, 0);
#endif
}

// ---- kernel 1: streaming f32 -> bf16 convert (RNE) --------------------
// Makes the 128MB fp32 weight set a 64MB bf16 set that FITS the 192MB L2,
// turning the GEMM phase compute-bound instead of HBM-bound.
__global__ __launch_bounds__(256)
void cvt_f32_to_bf16(const float* __restrict__ src,
                     unsigned int* __restrict__ dst,   // 2 bf16 per dword
                     int n4) {                         // n4 = elems/4
  const int i = blockIdx.x * 256 + threadIdx.x;
  if (i >= n4) return;
  const float4 v = ((const float4*)src)[i];
  uint2 d;
  d.x = pack_bf16_2(v.x, v.y);
  d.y = pack_bf16_2(v.z, v.w);
  ((uint2*)dst)[i] = d;
}

// ---- kernel 2: fused QLSTM GEMM + gates -------------------------------
// WG = 256 threads (8 wave32), tile 128 batch x 64 hidden, all 4 gates.
// Wave (rg, cg): rows rg*32..+32, cols cg*32..+32 -> 16 WMMA per K-slice.
// TDM double-buffers bf16 tiles from d_ws into LDS; s_wait_tensorcnt syncs.
__global__ __launch_bounds__(256)
void qlstm_fused_wmma(const unsigned short* __restrict__ Xb,
                      const unsigned short* __restrict__ Hb,
                      const unsigned short* __restrict__ Wib,
                      const unsigned short* __restrict__ Whb,
                      const float* __restrict__ c0,
                      const float* __restrict__ bhh,
                      float* __restrict__ out) {
  // bf16 pairs per dword: As[buf][row][k/2] (128x32k), Bs[buf][g*64+n][k/2]
  __shared__ __align__(16) unsigned int As[2][128 * 16];      // 2 x 8KB
  __shared__ __align__(16) unsigned int Bs[2][4 * 64 * 16];   // 2 x 16KB

  const int tid  = threadIdx.x;
  const int lane = tid & 31;
  const int wave = tid >> 5;
  const int rg   = wave & 3;           // row group (32 rows)
  const int cg   = wave >> 2;          // col group (32 cols)
  const int row0 = blockIdx.y * 128;
  const int col0 = blockIdx.x * 64;
  const int half = lane >> 4;
  const int l15  = lane & 15;

  const unsigned asA[2] = { lds_off(&As[0][0]), lds_off(&As[1][0]) };
  const unsigned asB[2] = { lds_off(&Bs[0][0]), lds_off(&Bs[1][0]) };
  const unsigned long long xsrc[2] = { (unsigned long long)Xb, (unsigned long long)Hb };
  const unsigned long long wsrc[2] = { (unsigned long long)Wib, (unsigned long long)Whb };

  // slice s in [0,128): phase = s>>6 (x*Wih then h0*Whh), kk = (s&63)*32
  auto issue = [&](int s, int buf) {
    const int ph = s >> 6;
    const int kk = (s & 63) * 32;
    // A: 2D tile 32(k) x 128(rows) of bf16 activations [4096 x 2048]
    tdm_load(xsrc[ph] + 2ull * ((unsigned long long)row0 * K_DIM + kk),
             asA[buf],
             /*tensor_d0*/ K_DIM, /*tensor_d1*/ B_DIM, /*tensor_d2*/ 0,
             /*tile*/ 32, 128, 0,
             /*stride0*/ K_DIM, /*stride1*/ 0);
    // B: 3D tile 32(k) x 64(n) x 4(gates) of bf16 weights [4][2048][2048]
    tdm_load(wsrc[ph] + 2ull * ((unsigned long long)col0 * K_DIM + kk),
             asB[buf],
             K_DIM, H_DIM, 4,
             32, 64, 4,
             K_DIM, (unsigned long long)H_DIM * K_DIM);
  };

  v8f acc[4][2][2] = {};               // [gate][m-subtile][n-subtile]

  if (wave == 0) issue(0, 0);

  for (int s = 0; s < 128; ++s) {
    const int cur = s & 1;
    if (wave == 0) {
      if (s + 1 < 128) {
        issue(s + 1, cur ^ 1);                   // prefetch next slice
        __builtin_amdgcn_s_wait_tensorcnt(2);    // current slice landed
      } else {
        __builtin_amdgcn_s_wait_tensorcnt(0);
      }
    }
    __syncthreads();

    const unsigned int* Ac = As[cur];
    const unsigned int* Bc = Bs[cur];

    // A frags (16x32 bf16): lanes 0-15 k0..7/16..23, lanes 16-31 k8..15/24..31
    FragBF a[2];
#pragma unroll
    for (int mi = 0; mi < 2; ++mi) {
      const int ar = (rg * 32 + mi * 16 + l15) * 16;
      a[mi].u[0] = *(const u32x4*)&Ac[ar + half * 4];
      a[mi].u[1] = *(const u32x4*)&Ac[ar + 8 + half * 4];
    }
#pragma unroll
    for (int g = 0; g < 4; ++g) {
#pragma unroll
      for (int t = 0; t < 2; ++t) {
        // B frag (32x16 bf16): lane = column, lanes 0-15 k0..7/16..23, 16-31 k8..15/24..31
        const int br = (g * 64 + cg * 32 + t * 16 + l15) * 16;
        FragBF b;
        b.u[0] = *(const u32x4*)&Bc[br + half * 4];
        b.u[1] = *(const u32x4*)&Bc[br + 8 + half * 4];
#pragma unroll
        for (int mi = 0; mi < 2; ++mi) {
          acc[g][mi][t] = __builtin_amdgcn_wmma_f32_16x16x32_bf16(
              false, a[mi].v, false, b.v, (short)0, acc[g][mi][t], false, false);
        }
      }
    }
    __syncthreads();
  }

  // ---- epilogue: C/D layout M = j + 8*(lane>=16), N = lane%16 ----
  const int mB = row0 + rg * 32 + half * 8;
  const int cB = col0 + cg * 32 + l15;
#pragma unroll
  for (int t = 0; t < 2; ++t) {
    const int col = cB + t * 16;
    const float bf = bhh[0 * H_DIM + col];
    const float bi = bhh[1 * H_DIM + col];
    const float bo = bhh[2 * H_DIM + col];
    const float bg = bhh[3 * H_DIM + col];
#pragma unroll
    for (int mi = 0; mi < 2; ++mi) {
      const int mbase = mB + mi * 16;
#pragma unroll
      for (int j = 0; j < 8; ++j) {
        const int m = mbase + j;
        const float fv = hsig(acc[0][mi][t][j] + bf);
        const float iv = hsig(acc[1][mi][t][j] + bi);
        const float ov = hsig(acc[2][mi][t][j] + bo);
        const float gv = htanh(acc[3][mi][t][j] + bg);
        const float c  = fv * c0[(size_t)m * H_DIM + col] + iv * gv;
        const float h  = ov * htanh(c);
        out[(size_t)m * H_DIM + col] = h;                              // h_1
        out[(size_t)B_DIM * H_DIM + (size_t)m * H_DIM + col] = c;      // c_1
      }
    }
  }
}

// ---- host launch ------------------------------------------------------
extern "C" void kernel_launch(void* const* d_in, const int* in_sizes, int n_in,
                              void* d_out, int out_size, void* d_ws, size_t ws_size,
                              hipStream_t stream) {
  (void)in_sizes; (void)n_in; (void)out_size; (void)ws_size;
  const float* x   = (const float*)d_in[0];   // input_  [B, IN]
  const float* h0  = (const float*)d_in[1];   // h_0     [B, H]
  const float* c0  = (const float*)d_in[2];   // c_0     [B, H]
  const float* Wih = (const float*)d_in[3];   // W_ih    [4H, IN]
  const float* Whh = (const float*)d_in[4];   // W_hh    [4H, H]
  const float* bhh = (const float*)d_in[5];   // b_hh    [4H]
  float* out = (float*)d_out;                 // h_1 then c_1, each [B, H]

  // bf16 staging layout in d_ws (96 MB total)
  unsigned short* Xb  = (unsigned short*)d_ws;                    // 8M elems
  unsigned short* Hb  = Xb  + (size_t)B_DIM * K_DIM;              // 8M elems
  unsigned short* Wib = Hb  + (size_t)B_DIM * H_DIM;              // 16M elems
  unsigned short* Whb = Wib + (size_t)4 * H_DIM * K_DIM;          // 16M elems

  const int nx4 = (B_DIM * K_DIM) / 4;          // 2M
  const int nw4 = (4 * H_DIM * K_DIM) / 4;      // 4M
  cvt_f32_to_bf16<<<nx4 / 256, 256, 0, stream>>>(x,   (unsigned int*)Xb,  nx4);
  cvt_f32_to_bf16<<<nx4 / 256, 256, 0, stream>>>(h0,  (unsigned int*)Hb,  nx4);
  cvt_f32_to_bf16<<<nw4 / 256, 256, 0, stream>>>(Wih, (unsigned int*)Wib, nw4);
  cvt_f32_to_bf16<<<nw4 / 256, 256, 0, stream>>>(Whh, (unsigned int*)Whb, nw4);

  dim3 grid(H_DIM / 64, B_DIM / 128);           // 32 x 32 workgroups
  qlstm_fused_wmma<<<grid, 256, 0, stream>>>(Xb, Hb, Wib, Whb, c0, bhh, out);
}
